// CATSynModel_73400991089148
// MI455X (gfx1250) — matrix operand order repaired and checked by reading, not compiled, not run on hardware
//
#include <hip/hip_runtime.h>
#include <hip/hip_bf16.h>

// MI455X (gfx1250) implementation.
// Compute-bound (~130 GFLOP GEMM vs ~11us of HBM traffic) -> bf16 WMMA
// (v_wmma_f32_16x16x32_bf16) with f32 accumulation for all dense GEMMs;
// f32 VALU + f32 global atomics for edge softmax / aggregation.
// Working set < 200 MB -> resident in the 192MB L2; no TDM staging needed.

typedef __bf16 bf16;
typedef __attribute__((ext_vector_type(16))) __bf16 v16bf;
typedef __attribute__((ext_vector_type(8)))  __bf16 v8bf;
typedef __attribute__((ext_vector_type(8)))  float  v8f;

#define N_NODES 8384
#define NE      50000
#define RELS    4
#define HEADS   4
#define OUT     256
#define FEATW   4096   // 16 slots * 256
#define AGGTOT  (N_NODES * FEATW)

// ---------------------------------------------------------------------------
// Generic WMMA GEMM: C[M,N] = A[M,K](bf16,row-major) x Bp(packed bf16) + bias
// block = 128 threads (4 waves); each wave computes 4 adjacent 16x16 N-tiles.
// grid = (M/16, N/256). Bp packed as [(K/32)][N/16][lane:32][e:16] bf16.
// ---------------------------------------------------------------------------
__global__ void gemm_wmma_bf16(const bf16* __restrict__ A,
                               const bf16* __restrict__ Bp,
                               const float* __restrict__ bias,
                               float* __restrict__ Cf,
                               bf16*  __restrict__ Cb,
                               int N, int K)
{
    const int lane = threadIdx.x & 31;
    const int wave = threadIdx.x >> 5;
    const int mt   = blockIdx.x;
    const int NT   = N >> 4;
    const int ntBase = (blockIdx.y * 4 + wave) * 4;
    if (ntBase >= NT) return;                       // wave-uniform

    const int row   = mt * 16 + (lane & 15);
    const int khalf = (lane >> 4) << 3;             // 0 or 8 per 16-bit A layout
    const bf16* __restrict__ Arow = A + (size_t)row * K;

    v8f acc0 = {}, acc1 = {}, acc2 = {}, acc3 = {};
    const int ksteps = K >> 5;
    for (int kt = 0; kt < ksteps; ++kt) {
        const int k0 = kt << 5;
        // 16-bit A 16x32 layout: lane<16 holds K {0..7,16..23}; lane>=16 {8..15,24..31}
        v8bf a_lo = *(const v8bf*)(Arow + k0 + khalf);
        v8bf a_hi = *(const v8bf*)(Arow + k0 + 16 + khalf);
        v16bf a;
        #pragma unroll
        for (int i = 0; i < 8; ++i) { a[i] = a_lo[i]; a[8 + i] = a_hi[i]; }

        const bf16* bpt = Bp + (((size_t)kt * NT + ntBase) * 32 + lane) * 16;
        v16bf b0 = *(const v16bf*)(bpt);
        v16bf b1 = *(const v16bf*)(bpt + 512);
        v16bf b2 = *(const v16bf*)(bpt + 1024);
        v16bf b3 = *(const v16bf*)(bpt + 1536);
        if (kt + 1 < ksteps) {                       // global_prefetch_b8
            __builtin_prefetch((const void*)(Arow + k0 + 32), 0, 0);
            __builtin_prefetch((const void*)(bpt + (size_t)NT * 512), 0, 0);
        }
        acc0 = __builtin_amdgcn_wmma_f32_16x16x32_bf16(false, a, false, b0, (short)0, acc0, false, false);
        acc1 = __builtin_amdgcn_wmma_f32_16x16x32_bf16(false, a, false, b1, (short)0, acc1, false, false);
        acc2 = __builtin_amdgcn_wmma_f32_16x16x32_bf16(false, a, false, b2, (short)0, acc2, false, false);
        acc3 = __builtin_amdgcn_wmma_f32_16x16x32_bf16(false, a, false, b3, (short)0, acc3, false, false);
    }

    // f32 C/D layout: VGPR r -> (M = r + 8*(lane>=16), N = lane&15)
    const int colLocal = lane & 15;
    const int rbase    = mt * 16 + ((lane >> 4) << 3);
    v8f accs[4] = {acc0, acc1, acc2, acc3};
    #pragma unroll
    for (int q = 0; q < 4; ++q) {
        int col = (ntBase + q) * 16 + colLocal;
        float bval = bias ? bias[col] : 0.0f;
        #pragma unroll
        for (int r = 0; r < 8; ++r) {
            float v = accs[q][r] + bval;
            size_t idx = (size_t)(rbase + r) * N + col;
            if (Cf) Cf[idx] = v;
            if (Cb) Cb[idx] = (bf16)v;
        }
    }
}

// Pack a row-major f32 [K,Ncols] (or its transpose) into the WMMA-B bf16 layout
// of a GEMM whose full width is NTtotal*16, starting at column tile ntOffset.
__global__ void pack_b(const float* __restrict__ src, bf16* __restrict__ dst,
                       int K, int Ncols, int ldsrc, int transpose,
                       int NTtotal, int ntOffset)
{
    int idx = blockIdx.x * blockDim.x + threadIdx.x;
    int total = K * Ncols;
    if (idx >= total) return;
    int e    = idx & 15;
    int lane = (idx >> 4) & 31;
    int t2   = idx >> 9;
    int NTl  = Ncols >> 4;
    int ntL  = t2 % NTl;
    int kt   = t2 / NTl;
    int k = kt * 32 + ((lane >> 4) << 3) + (e & 7) + ((e >> 3) << 4);   // lane>=16 -> K + 8; e>=8 -> K + 16
    int n = ntL * 16 + (lane & 15);
    float v = transpose ? src[(size_t)n * ldsrc + k] : src[(size_t)k * ldsrc + n];
    dst[(((size_t)kt * NTtotal + (ntOffset + ntL)) * 32 + lane) * 16 + e] = (bf16)v;
}

// ---------------------------------------------------------------------------
// Small dense helpers (tiny matrices; VALU is fine)
// ---------------------------------------------------------------------------
__global__ void lin_f32(const float* __restrict__ A, const float* __restrict__ B,
                        const float* __restrict__ b, float* __restrict__ C,
                        int M, int N, int K)
{
    int idx = blockIdx.x * blockDim.x + threadIdx.x;
    if (idx >= M * N) return;
    int m = idx / N, n = idx % N;
    float s = b[n];
    const float* a = A + (size_t)m * K;
    for (int k = 0; k < K; ++k) s += a[k] * B[(size_t)k * N + n];
    C[idx] = s;
}

// fused[k][n] = sum_j A[k][j] * P[pOff+n][j]     (fold MHA in-proj into wq/wk/wv)
__global__ void fuse_w(const float* __restrict__ A, const float* __restrict__ P,
                       int pOff, float* __restrict__ out, int Ka)
{
    int idx = blockIdx.x * blockDim.x + threadIdx.x;
    if (idx >= Ka * 256) return;
    int k = idx >> 8, n = idx & 255;
    const float* a = A + (size_t)k * 256;
    const float* p = P + (size_t)(pOff + n) * 256;
    float s = 0.f;
    for (int j = 0; j < 256; ++j) s += a[j] * p[j];
    out[idx] = s;
}

__global__ void fuse_b(const float* __restrict__ bvec, const float* __restrict__ P,
                       const float* __restrict__ ipb, int pOff, float* __restrict__ out)
{
    int n = blockIdx.x * blockDim.x + threadIdx.x;
    if (n >= 256) return;
    const float* p = P + (size_t)(pOff + n) * 256;
    float s = ipb[pOff + n];
    for (int j = 0; j < 256; ++j) s += bvec[j] * p[j];
    out[n] = s;
}

__global__ void cvec_mean(const float* __restrict__ cfv, float* __restrict__ cvm)
{
    int j = threadIdx.x;                       // 128 threads
    float s = 0.f;
    for (int i = 0; i < 64; ++i) s += cfv[i * 128 + j];
    cvm[j] = s * (1.0f / 64.0f);
}

// ---------------------------------------------------------------------------
// Node-feature builders
// ---------------------------------------------------------------------------
__global__ void build_inf1(const float* __restrict__ x, const float* __restrict__ duc,
                           const float* __restrict__ cfv, const float* __restrict__ cvm,
                           bf16* __restrict__ infB, bf16* __restrict__ cfinB)
{
    int n = blockIdx.x;
    const float* dsrc = (n < 8192) ? x + (size_t)(n & 127) * 256
                      : (n < 8256) ? duc + (size_t)(n - 8192) * 256
                                   : x + (size_t)(n - 8256) * 256;
    for (int c = threadIdx.x; c < 256; c += blockDim.x)
        infB[(size_t)n * 384 + c] = (bf16)dsrc[c];
    for (int c = threadIdx.x; c < 128; c += blockDim.x) {
        float v = (n < 8192) ? cfv[(size_t)(n >> 7) * 128 + c]
                : (n < 8256) ? cfv[(size_t)(n - 8192) * 128 + c]
                             : cvm[c];
        infB[(size_t)n * 384 + 256 + c] = (bf16)v;
        cfinB[(size_t)n * 128 + c] = (bf16)v;
    }
}

__global__ void build_inf2(const bf16* __restrict__ hB, const bf16* __restrict__ cfinB,
                           bf16* __restrict__ infB)
{
    int n = blockIdx.x;
    for (int c = threadIdx.x; c < 256; c += blockDim.x)
        infB[(size_t)n * 384 + c] = hB[(size_t)n * 256 + c];
    for (int c = threadIdx.x; c < 128; c += blockDim.x)
        infB[(size_t)n * 384 + 256 + c] = cfinB[(size_t)n * 128 + c];
}

// ---------------------------------------------------------------------------
// GAT attention pieces
// ---------------------------------------------------------------------------
__global__ void el_er(const bf16* __restrict__ featB, const float* __restrict__ al,
                      const float* __restrict__ ar, float* __restrict__ el,
                      float* __restrict__ er)
{
    int gwv  = (blockIdx.x * blockDim.x + threadIdx.x) >> 5;  // wave per (n, slot)
    int lane = threadIdx.x & 31;
    if (gwv >= N_NODES * 16) return;
    int s = gwv & 15;
    const bf16*  f  = featB + (size_t)(gwv >> 4) * FEATW + (size_t)s * 256;
    const float* aL = al + (size_t)s * 256;
    const float* aR = ar + (size_t)s * 256;
    float sl = 0.f, sr = 0.f;
    for (int i = lane; i < 256; i += 32) {
        float fv = (float)f[i];
        sl += fv * aL[i];
        sr += fv * aR[i];
    }
    for (int o = 16; o; o >>= 1) { sl += __shfl_down(sl, o); sr += __shfl_down(sr, o); }
    if (!lane) { el[gwv] = sl; er[gwv] = sr; }
}

__device__ __forceinline__ unsigned fmap(float f) {
    unsigned u = __float_as_uint(f);
    return (u & 0x80000000u) ? ~u : (u | 0x80000000u);   // monotonic float->uint
}

__global__ void fill_zero_f32(float* p, int n) {
    int i = blockIdx.x * blockDim.x + threadIdx.x;
    if (i < n) p[i] = 0.f;
}
__global__ void fill_zero_u32(unsigned* p, int n) {
    int i = blockIdx.x * blockDim.x + threadIdx.x;
    if (i < n) p[i] = 0u;
}

__global__ void edge_max(const int* __restrict__ edges, const float* __restrict__ el,
                         const float* __restrict__ er, unsigned* __restrict__ mu)
{
    int idx = blockIdx.x * blockDim.x + threadIdx.x;    // (t, edge, head)
    if (idx >= RELS * NE * HEADS) return;
    int h = idx & 3, ei = (idx >> 2) % NE, t = (idx >> 2) / NE;
    int src = edges[t * 2 * NE + ei];
    int dst = edges[t * 2 * NE + NE + ei];
    int sl = t * 4 + h;
    float e = el[src * 16 + sl] + er[dst * 16 + sl];
    e = (e > 0.f) ? e : 0.2f * e;
    atomicMax(&mu[((size_t)t * N_NODES + dst) * 4 + h], fmap(e));
}

__global__ void finalize_m(const unsigned* __restrict__ mu, float* __restrict__ mf, int n)
{
    int i = blockIdx.x * blockDim.x + threadIdx.x;
    if (i >= n) return;
    unsigned u = mu[i];
    float v = 0.f;
    if (u != 0u) {
        unsigned b = (u & 0x80000000u) ? (u & 0x7FFFFFFFu) : ~u;
        v = __uint_as_float(b);
        if (!isfinite(v)) v = 0.f;                      // jnp.where(isfinite(m), m, 0)
    }
    mf[i] = v;
}

__global__ void edge_sum(const int* __restrict__ edges, const float* __restrict__ el,
                         const float* __restrict__ er, const float* __restrict__ mf,
                         float* __restrict__ sf)
{
    int idx = blockIdx.x * blockDim.x + threadIdx.x;
    if (idx >= RELS * NE * HEADS) return;
    int h = idx & 3, ei = (idx >> 2) % NE, t = (idx >> 2) / NE;
    int src = edges[t * 2 * NE + ei];
    int dst = edges[t * 2 * NE + NE + ei];
    int sl = t * 4 + h;
    float e = el[src * 16 + sl] + er[dst * 16 + sl];
    e = (e > 0.f) ? e : 0.2f * e;
    float p = __expf(e - mf[((size_t)t * N_NODES + dst) * 4 + h]);
    atomicAdd(&sf[((size_t)t * N_NODES + dst) * 4 + h], p);
}

__global__ void edge_agg(const int* __restrict__ edges, const float* __restrict__ el,
                         const float* __restrict__ er, const float* __restrict__ mf,
                         const float* __restrict__ sf, const bf16* __restrict__ featB,
                         float* __restrict__ agg)
{
    int gwv  = (blockIdx.x * blockDim.x + threadIdx.x) >> 5;  // wave per (t,edge,head)
    int lane = threadIdx.x & 31;
    if (gwv >= RELS * NE * HEADS) return;
    int h = gwv & 3, ei = (gwv >> 2) % NE, t = (gwv >> 2) / NE;
    int src = edges[t * 2 * NE + ei];
    int dst = edges[t * 2 * NE + NE + ei];
    int sl = t * 4 + h;
    float e = el[src * 16 + sl] + er[dst * 16 + sl];
    e = (e > 0.f) ? e : 0.2f * e;
    size_t ms = ((size_t)t * N_NODES + dst) * 4 + h;
    float a = __expf(e - mf[ms]) / fmaxf(sf[ms], 1e-9f);
    const bf16* f = featB + (size_t)src * FEATW + (size_t)sl * 256;
    float*      o = agg   + (size_t)dst * FEATW + (size_t)sl * 256;
    for (int d = lane; d < 256; d += 32)
        atomicAdd(&o[d], a * (float)f[d]);
}

__global__ void bias_cvt(const float* __restrict__ agg, const float* __restrict__ gb,
                         bf16* __restrict__ rstB, int total)
{
    int i = blockIdx.x * blockDim.x + threadIdx.x;
    if (i >= total) return;
    rstB[i] = (bf16)(agg[i] + gb[i & (FEATW - 1)]);
}

// ---------------------------------------------------------------------------
// Per-node MHA: single head, 16 keys, E=256. One block (256 thr) per node.
// ---------------------------------------------------------------------------
__global__ void attn(const float* __restrict__ qf, const bf16* __restrict__ kB,
                     const bf16* __restrict__ vB, bf16* __restrict__ oB)
{
    int n = blockIdx.x, d = threadIdx.x;
    __shared__ float qs[256];
    __shared__ float red[256];
    __shared__ float sc[16];
    qs[d] = qf[(size_t)n * 256 + d];
    __syncthreads();
    for (int j = 0; j < 16; ++j) {
        red[d] = qs[d] * (float)kB[((size_t)n * 16 + j) * 256 + d];
        __syncthreads();
        for (int st = 128; st; st >>= 1) {
            if (d < st) red[d] += red[d + st];
            __syncthreads();
        }
        if (!d) sc[j] = red[0] * (1.0f / 16.0f);        // / sqrt(256)
        __syncthreads();
    }
    float mx = -1e30f;
    for (int j = 0; j < 16; ++j) mx = fmaxf(mx, sc[j]);
    float w[16], sum = 0.f;
    for (int j = 0; j < 16; ++j) { w[j] = __expf(sc[j] - mx); sum += w[j]; }
    float inv = 1.0f / sum, o = 0.f;
    for (int j = 0; j < 16; ++j)
        o += w[j] * inv * (float)vB[((size_t)n * 16 + j) * 256 + d];
    oB[(size_t)n * 256 + d] = (bf16)o;
}

// ---------------------------------------------------------------------------
extern "C" void kernel_launch(void* const* d_in, const int* in_sizes, int n_in,
                              void* d_out, int out_size, void* d_ws, size_t ws_size,
                              hipStream_t stream)
{
    (void)in_sizes; (void)n_in; (void)out_size; (void)ws_size;

    const float* x    = (const float*)d_in[0];
    const float* cfm  = (const float*)d_in[1];
    const float* cfnw = (const float*)d_in[2];
    const float* cfnb = (const float*)d_in[3];
    const float* htw  = (const float*)d_in[4];
    const float* htb  = (const float*)d_in[5];
    const int*   edges = (const int*)d_in[34];

    // -------- workspace carve (~235 MB, with buffer aliasing) --------
    char* w = (char*)d_ws;
    size_t off = 0;
    auto carve = [&](size_t bytes) -> char* {
        char* p = w + off;
        off = (off + bytes + 255) & ~(size_t)255;
        return p;
    };
    float* cfvec = (float*)carve(64 * 128 * 4);
    float* duc   = (float*)carve(64 * 256 * 4);
    float* cvm   = (float*)carve(128 * 4);
    bf16*  cfinB = (bf16*)carve((size_t)N_NODES * 128 * 2);
    bf16*  infB  = (bf16*)carve((size_t)N_NODES * 384 * 2);
    bf16*  big1  = (bf16*)carve((size_t)AGGTOT * 2);      // featB -> rstB
    float* agg   = (float*)carve((size_t)AGGTOT * 4);     // agg f32 -> (kB|vB) bf16
    bf16*  kB    = (bf16*)agg;
    bf16*  vB    = kB + (size_t)N_NODES * 16 * 256;
    float* el    = (float*)carve((size_t)N_NODES * 16 * 4);
    float* er    = (float*)carve((size_t)N_NODES * 16 * 4);
    unsigned* mu = (unsigned*)carve((size_t)RELS * N_NODES * 4 * 4);
    float* mf    = (float*)carve((size_t)RELS * N_NODES * 4 * 4);
    float* sf    = (float*)carve((size_t)RELS * N_NODES * 4 * 4);
    float* qf    = (float*)carve((size_t)N_NODES * 256 * 4);
    bf16*  oB    = (bf16*)carve((size_t)N_NODES * 256 * 2);
    bf16*  hB    = (bf16*)carve((size_t)N_NODES * 256 * 2);
    float* tmpW  = (float*)carve(256 * 256 * 4);
    bf16*  gwp[2]; bf16* wqp[2]; bf16* wkp[2]; bf16* wvp[2]; bf16* wopp[2];
    float* bq[2]; float* bk[2]; float* bv[2];
    for (int l = 0; l < 2; ++l) {
        gwp[l]  = (bf16*)carve((size_t)384 * 4096 * 2);
        wqp[l]  = (bf16*)carve(128 * 256 * 2);
        wkp[l]  = (bf16*)carve(256 * 256 * 2);
        wvp[l]  = (bf16*)carve(256 * 256 * 2);
        wopp[l] = (bf16*)carve(256 * 256 * 2);
        bq[l] = (float*)carve(256 * 4);
        bk[l] = (float*)carve(256 * 4);
        bv[l] = (float*)carve(256 * 4);
    }

    // -------- weight preparation (fold MHA in-proj, pack to WMMA-B) --------
    for (int l = 0; l < 2; ++l) {
        int b = 6 + 14 * l;
        const float* gw   = (const float*)d_in[b + 0];
        const float* wq_w = (const float*)d_in[b + 4];
        const float* wq_b = (const float*)d_in[b + 5];
        const float* wk_w = (const float*)d_in[b + 6];
        const float* wk_b = (const float*)d_in[b + 7];
        const float* wv_w = (const float*)d_in[b + 8];
        const float* wv_b = (const float*)d_in[b + 9];
        const float* ip_w = (const float*)d_in[b + 10];
        const float* ip_b = (const float*)d_in[b + 11];
        const float* op_w = (const float*)d_in[b + 12];

        fuse_w<<<(128 * 256 + 255) / 256, 256, 0, stream>>>(wq_w, ip_w, 0, tmpW, 128);
        fuse_b<<<1, 256, 0, stream>>>(wq_b, ip_w, ip_b, 0, bq[l]);
        pack_b<<<(128 * 256 + 255) / 256, 256, 0, stream>>>(tmpW, wqp[l], 128, 256, 256, 0, 16, 0);

        fuse_w<<<(256 * 256 + 255) / 256, 256, 0, stream>>>(wk_w, ip_w, 256, tmpW, 256);
        fuse_b<<<1, 256, 0, stream>>>(wk_b, ip_w, ip_b, 256, bk[l]);
        pack_b<<<(256 * 256 + 255) / 256, 256, 0, stream>>>(tmpW, wkp[l], 256, 256, 256, 0, 16, 0);

        fuse_w<<<(256 * 256 + 255) / 256, 256, 0, stream>>>(wv_w, ip_w, 512, tmpW, 256);
        fuse_b<<<1, 256, 0, stream>>>(wv_b, ip_w, ip_b, 512, bv[l]);
        pack_b<<<(256 * 256 + 255) / 256, 256, 0, stream>>>(tmpW, wvp[l], 256, 256, 256, 0, 16, 0);

        pack_b<<<(256 * 256 + 255) / 256, 256, 0, stream>>>(op_w, wopp[l], 256, 256, 256, 1, 16, 0);

        for (int t = 0; t < RELS; ++t)
            pack_b<<<(384 * 1024 + 255) / 256, 256, 0, stream>>>(
                gw + (size_t)t * 384 * 1024, gwp[l], 384, 1024, 1024, 0, 256, t * 64);
    }

    // -------- node-feature preprocessing --------
    lin_f32<<<(64 * 128 + 255) / 256, 256, 0, stream>>>(cfm, cfnw, cfnb, cfvec, 64, 128, 512);
    lin_f32<<<(64 * 256 + 255) / 256, 256, 0, stream>>>(cfm, htw, htb, duc, 64, 256, 512);
    cvec_mean<<<1, 128, 0, stream>>>(cfvec, cvm);
    build_inf1<<<N_NODES, 128, 0, stream>>>(x, duc, cfvec, cvm, infB, cfinB);

    // -------- two CAGAT layers --------
    for (int l = 0; l < 2; ++l) {
        int b = 6 + 14 * l;
        const float* alp = (const float*)d_in[b + 1];
        const float* arp = (const float*)d_in[b + 2];
        const float* gbp = (const float*)d_in[b + 3];
        const float* opb = (const float*)d_in[b + 13];

        // feat = inf @ gw (4 relations fused into one N=4096 GEMM), no bias
        gemm_wmma_bf16<<<dim3(N_NODES / 16, 16), 128, 0, stream>>>(
            infB, gwp[l], nullptr, nullptr, big1, 4096, 384);

        el_er<<<(N_NODES * 16 * 32 + 255) / 256, 256, 0, stream>>>(big1, alp, arp, el, er);

        fill_zero_f32<<<(AGGTOT + 255) / 256, 256, 0, stream>>>(agg, AGGTOT);
        fill_zero_u32<<<(RELS * N_NODES * 4 + 255) / 256, 256, 0, stream>>>(mu, RELS * N_NODES * 4);
        fill_zero_f32<<<(RELS * N_NODES * 4 + 255) / 256, 256, 0, stream>>>(sf, RELS * N_NODES * 4);

        edge_max<<<(RELS * NE * HEADS + 255) / 256, 256, 0, stream>>>(edges, el, er, mu);
        finalize_m<<<(RELS * N_NODES * 4 + 255) / 256, 256, 0, stream>>>(mu, mf, RELS * N_NODES * 4);
        edge_sum<<<(RELS * NE * HEADS + 255) / 256, 256, 0, stream>>>(edges, el, er, mf, sf);
        edge_agg<<<(RELS * NE * HEADS * 32 + 255) / 256, 256, 0, stream>>>(
            edges, el, er, mf, sf, big1, agg);

        // rst = agg + gb -> bf16 (overwrites featB; feat no longer needed)
        bias_cvt<<<(AGGTOT + 255) / 256, 256, 0, stream>>>(agg, gbp, big1, AGGTOT);

        // K/V: [134144,256] @ [256,256] (+fused bias) -> bf16 into old agg space
        gemm_wmma_bf16<<<dim3(N_NODES * 16 / 16, 1), 128, 0, stream>>>(
            big1, wkp[l], bk[l], nullptr, kB, 256, 256);
        gemm_wmma_bf16<<<dim3(N_NODES * 16 / 16, 1), 128, 0, stream>>>(
            big1, wvp[l], bv[l], nullptr, vB, 256, 256);
        // Q: [8384,128] @ [128,256] (+fused bias) -> f32
        gemm_wmma_bf16<<<dim3(N_NODES / 16, 1), 128, 0, stream>>>(
            cfinB, wqp[l], bq[l], qf, nullptr, 256, 128);

        attn<<<N_NODES, 256, 0, stream>>>(qf, kB, vB, oB);

        // out-proj: o @ op_w^T + op_b
        if (l == 0) {
            gemm_wmma_bf16<<<dim3(N_NODES / 16, 1), 128, 0, stream>>>(
                oB, wopp[l], opb, nullptr, hB, 256, 256);
            build_inf2<<<N_NODES, 128, 0, stream>>>(hB, cfinB, infB);
        } else {
            gemm_wmma_bf16<<<dim3(N_NODES / 16, 1), 128, 0, stream>>>(
                oB, wopp[l], opb, (float*)d_out, nullptr, 256, 256);
        }
    }
}
